// STALSTM_84447646974548
// MI455X (gfx1250) — matrix operand order repaired and checked by reading, not compiled, not run on hardware
//
#include <hip/hip_runtime.h>
#include <hip/hip_bf16.h>

// ---------------------------------------------------------------------------
// STA-LSTM for MI455X (gfx1250), wave32 + WMMA bf16 16x16x32, f32 accumulate.
//
//   1. convert weights fp32 -> bf16 (W_ih K padded 25->32 for the J-input layers)
//   2. pack x (optionally * alpha) into padded bf16 A matrix [T*B, 32]
//   3. Gx = X @ W_ih^T + bias   (WMMA GEMM, double-buffered LDS tiles filled
//      with GLOBAL_LOAD_ASYNC_TO_LDS_B128, waited with s_wait_asynccnt)
//   4. persistent recurrent kernel: 16 WGs/direction gang-step through T,
//      c-state register resident, h double-buffered in global (L2),
//      one device barrier per step
//   5. attention (fc1/tanh/fc2/softmax/top-5/floor) -> alpha
//   6. repeat 2-4 for main layers 0..2, then final FC from hn_main
// ---------------------------------------------------------------------------

typedef __bf16 bf16_t;
typedef __attribute__((ext_vector_type(16))) __bf16 bf16x16;
typedef __attribute__((ext_vector_type(8)))  __bf16 bf16x8;
typedef __attribute__((ext_vector_type(8)))  float  f32x8;

static constexpr int kT  = 256;   // seq len
static constexpr int kB  = 128;   // batch
static constexpr int kJ  = 25;    // joints
static constexpr int kH  = 256;   // hidden
static constexpr int kG  = 4 * kH;        // 1024 gate cols
static constexpr int kTB = kT * kB;       // 32768 rows
static constexpr int RNWG = 16;           // workgroups per direction in recurrence

// ---------------- WMMA fragment helpers (ISA 7.12.2 layouts) ----------------

__device__ inline bf16x16 load_frag_a(const bf16_t* tile, int ld) {
  // A: 16x32 bf16, row-major source. lane<16: row=lane, k = 0..7 & 16..23
  //                                  lane>=16: row=lane-16, k = 8..15 & 24..31
  int lane = threadIdx.x & 31;
  const bf16_t* p = tile + (size_t)(lane & 15) * ld + ((lane >> 4) << 3);
  union { bf16x16 v; bf16x8 h[2]; } u;
  u.h[0] = *(const bf16x8*)(p);
  u.h[1] = *(const bf16x8*)(p + 16);
  return u.v;
}

__device__ inline bf16x16 load_frag_b(const bf16_t* wrow, int ld) {
  // B = W^T (W row-major [N][K]). lane<16: col n=lane holds k=0..7 & 16..23,
  // lane>=16: col n=lane-16 holds k=8..15 & 24..31.
  int lane = threadIdx.x & 31;
  const bf16_t* p = wrow + (size_t)(lane & 15) * ld + ((lane >> 4) << 3);
  union { bf16x16 v; bf16x8 h[2]; } u;
  u.h[0] = *(const bf16x8*)(p);
  u.h[1] = *(const bf16x8*)(p + 16);
  return u.v;
}

__device__ inline f32x8 wmma_bf16(bf16x16 a, bf16x16 b, f32x8 c) {
  return __builtin_amdgcn_wmma_f32_16x16x32_bf16(false, a, false, b,
                                                 (short)0, c, false, false);
}

__device__ inline float sigmoidf_(float x) { return 1.f / (1.f + __expf(-x)); }

// ---------------- gfx1250 async memory->LDS path ----------------------------
// GLOBAL_LOAD_ASYNC_TO_LDS_B128: 16B global -> LDS, tracked by ASYNCcnt
// (CDNA5 ISA). Generic pointers to __shared__ carry the
// wave-relative LDS byte offset in their low 32 bits (§10.2 aperture map).

__device__ inline unsigned lds_off_of(const void* p) {
  return (unsigned)(unsigned long long)p;
}

__device__ inline void async_load16(unsigned lds_off, const void* gaddr) {
  asm volatile("global_load_async_to_lds_b128 %0, %1, off"
               :: "v"(lds_off), "v"((unsigned long long)gaddr)
               : "memory");
}

__device__ inline void wait_async0() {
  asm volatile("s_wait_asynccnt 0" ::: "memory");
}

// ---------------- device-scope gang barrier (monotonic counter) -------------

__device__ inline void gang_barrier(unsigned* bar, unsigned target) {
  __threadfence();           // make this WG's global stores visible
  __syncthreads();
  if (threadIdx.x == 0) {
    atomicAdd(bar, 1u);
    while (__hip_atomic_load(bar, __ATOMIC_RELAXED,
                             __HIP_MEMORY_SCOPE_AGENT) < target)
      __builtin_amdgcn_s_sleep(1);
  }
  __syncthreads();
  __threadfence();           // acquire side
}

// ---------------- weight convert / pad (fp32 -> bf16) -----------------------

__global__ void k_cvt(const float* __restrict__ src, bf16_t* __restrict__ dst,
                      int rows, int kin, int kout) {
  int i = blockIdx.x * blockDim.x + threadIdx.x;
  if (i >= rows * kout) return;
  int r = i / kout, k = i - r * kout;
  dst[i] = (bf16_t)((k < kin) ? src[(size_t)r * kin + k] : 0.f);
}

// ---------------- pack x (optionally * alpha) into padded bf16 A ------------

__global__ void k_pack25(const float* __restrict__ x, const float* __restrict__ alpha,
                         bf16_t* __restrict__ out) {
  int i = blockIdx.x * blockDim.x + threadIdx.x;
  if (i >= kTB * 32) return;
  int row = i >> 5, k = i & 31;
  float v = 0.f;
  if (k < kJ) {
    v = x[(size_t)row * kJ + k];
    if (alpha) v *= alpha[(row % kB) * kJ + k];
  }
  out[i] = (bf16_t)v;
}

// ---------------- Gx GEMM: G[dir] = A @ W[dir]^T + bias[dir] ----------------
// A: [kTB][K] bf16 row-major, W: [kG][K] bf16 row-major, G: [kTB][kG] fp32.
// Block tile 128(M) x 64(N), 8 waves, wave = one 16-row strip x 4 N-tiles.
// A tile (128x32) and W tile (64x32) are double-buffered in LDS and filled
// with async global->LDS loads that overlap the WMMA work on the other buffer.

__global__ void k_gates_gemm(const bf16_t* __restrict__ A,
                             const bf16_t* __restrict__ Wf,
                             const bf16_t* __restrict__ Wb,
                             const float* __restrict__ biasf,
                             const float* __restrict__ biasb,
                             float* __restrict__ Gx, int K) {
  __shared__ bf16_t As[2][128][40];  // 40-elem pitch = 80B = 5x16B, bank-spread
  __shared__ bf16_t Ws[2][64][40];
  const bf16_t* W   = blockIdx.z ? Wb : Wf;
  const float* bias = blockIdx.z ? biasb : biasf;
  float* G = Gx + (size_t)blockIdx.z * (size_t)kTB * kG;

  const int m0 = blockIdx.y * 128;
  const int n0 = blockIdx.x * 64;
  const int wave = threadIdx.x >> 5, lane = threadIdx.x & 31;
  const int tid = threadIdx.x;

  // staging coordinates: A = 128 rows x 32 elems (2 thr/row, 16 elems each),
  //                      W = 64 rows x 32 elems (4 thr/row, 8 elems each)
  const int ar = tid >> 1, ach = (tid & 1) << 4;
  const int wr = tid >> 2, wch = (tid & 3) << 3;

  auto stage = [&](int k0, int buf) {
    const bf16_t* asrc = A + (size_t)(m0 + ar) * K + k0 + ach;
    async_load16(lds_off_of(&As[buf][ar][ach]),     asrc);
    async_load16(lds_off_of(&As[buf][ar][ach + 8]), asrc + 8);
    const bf16_t* wsrc = W + (size_t)(n0 + wr) * K + k0 + wch;
    async_load16(lds_off_of(&Ws[buf][wr][wch]),     wsrc);
  };

  f32x8 acc[4] = {};
  stage(0, 0);
  wait_async0();
  __syncthreads();

  const int nk = K >> 5;
  for (int i = 0; i < nk; ++i) {
    const int cur = i & 1;
    if (i + 1 < nk) stage((i + 1) << 5, cur ^ 1);   // prefetch next chunk

    bf16x16 af = load_frag_a(&As[cur][wave * 16][0], 40);
#pragma unroll
    for (int j = 0; j < 4; ++j) {
      bf16x16 bf = load_frag_b(&Ws[cur][j * 16][0], 40);
      acc[j] = wmma_bf16(af, bf, acc[j]);
    }
    wait_async0();     // own async loads for next buffer done
    __syncthreads();   // everyone's loads + reads of `cur` done
  }

  int n = lane & 15, moff = (lane >> 4) << 3;
#pragma unroll
  for (int j = 0; j < 4; ++j)
#pragma unroll
    for (int r = 0; r < 8; ++r) {
      int gm = m0 + wave * 16 + moff + r;
      int gn = n0 + j * 16 + n;
      G[(size_t)gm * kG + gn] = acc[j][r] + bias[gn];
    }
}

// ---------------- persistent bi-directional LSTM recurrence -----------------
// grid (RNWG, 2): blockIdx.y = direction. Each WG owns 16 hidden columns,
// each of its 8 waves owns 16 batch rows -> c state is a v8f register tile.

__global__ void k_lstm_recur(const float* __restrict__ Gx,          // [2][kTB][kG]
                             const bf16_t* __restrict__ whh_f,
                             const bf16_t* __restrict__ whh_b,      // [kG][kH]
                             const float* __restrict__ h0,
                             const float* __restrict__ c0,          // [2][kB][kH]
                             bf16_t* __restrict__ hbuf,             // [2][2][kB][kH]
                             bf16_t* __restrict__ y,                // [kT][kB][2H] | null
                             float* __restrict__ hn,                // [2][kB][kH]
                             unsigned* __restrict__ bar) {          // [2]
  const int dir   = blockIdx.y;
  const int wg    = blockIdx.x;
  const int wave  = threadIdx.x >> 5, lane = threadIdx.x & 31;
  const int ncol0 = wg * 16;                // hidden column base
  const int b0    = wave * 16;              // batch row base
  const int n     = lane & 15, moff = (lane >> 4) << 3;

  const bf16_t* whh = dir ? whh_b : whh_f;
  const float*  gx  = Gx + (size_t)dir * kTB * kG;
  const float*  h0d = h0 + (size_t)dir * kB * kH;
  const float*  c0d = c0 + (size_t)dir * kB * kH;
  bf16_t*       hb  = hbuf + (size_t)dir * 2 * kB * kH;
  unsigned*     mybar = bar + dir;

  // c-state lives in registers for the whole sequence
  f32x8 creg;
#pragma unroll
  for (int r = 0; r < 8; ++r)
    creg[r] = c0d[(size_t)(b0 + moff + r) * kH + ncol0 + n];

  // initialize h double-buffer slot 0 with h0 (this WG's column slice)
  for (int e = threadIdx.x; e < kB * 16; e += blockDim.x) {
    int row = e >> 4, col = e & 15;
    hb[(size_t)row * kH + ncol0 + col] =
        (bf16_t)h0d[(size_t)row * kH + ncol0 + col];
  }
  gang_barrier(mybar, RNWG * 1u);

  for (int t = 0; t < kT; ++t) {
    const int tdata = dir ? (kT - 1 - t) : t;
    const bf16_t* hcur = hb + (size_t)(t & 1) * kB * kH;
    bf16_t*       hnxt = hb + (size_t)((t + 1) & 1) * kB * kH;

    f32x8 acc[4] = {};
#pragma unroll
    for (int kk = 0; kk < 8; ++kk) {
      bf16x16 af = load_frag_a(hcur + (size_t)b0 * kH + kk * 32, kH);
#pragma unroll
      for (int s = 0; s < 4; ++s) {  // gate strips i,f,g,o
        bf16x16 bf = load_frag_b(whh + (size_t)(s * kH + ncol0) * kH + kk * 32, kH);
        acc[s] = wmma_bf16(af, bf, acc[s]);
      }
    }

    const float* gxrow0 = gx + (size_t)tdata * kB * kG;
    if (t + 1 < kT) {   // pull next step's Gx tile toward the caches
      int tn = dir ? (kT - 2 - t) : (t + 1);
      __builtin_prefetch(gx + ((size_t)tn * kB + b0 + moff) * kG + ncol0 + n, 0, 1);
    }

#pragma unroll
    for (int r = 0; r < 8; ++r) {
      int b   = b0 + moff + r;
      int col = ncol0 + n;
      const float* grow = gxrow0 + (size_t)b * kG;
      float gi = acc[0][r] + grow[col];
      float gf = acc[1][r] + grow[kH + col];
      float gg = acc[2][r] + grow[2 * kH + col];
      float go = acc[3][r] + grow[3 * kH + col];
      float cc = sigmoidf_(gf) * creg[r] + sigmoidf_(gi) * tanhf(gg);
      creg[r]  = cc;
      float hh = sigmoidf_(go) * tanhf(cc);
      hnxt[(size_t)b * kH + col] = (bf16_t)hh;
      if (y) y[((size_t)tdata * kB + b) * (2 * kH) + dir * kH + col] = (bf16_t)hh;
      if (t == kT - 1)
        hn[(size_t)dir * kB * kH + (size_t)b * kH + col] = hh;
    }
    gang_barrier(mybar, RNWG * (unsigned)(t + 2));
  }
}

// ---------------- attention: softmax(fc2(tanh(fc1(mean(hn))))) + top-5 ------

__global__ void k_attn(const float* __restrict__ hn_sa,   // [2][kB][kH]
                       const float* __restrict__ fc1w, const float* __restrict__ fc1b,
                       const float* __restrict__ fc2w, const float* __restrict__ fc2b,
                       float* __restrict__ alpha) {       // [kB][kJ]
  __shared__ float hm[kH], z1[kH], lg[32];
  int b = blockIdx.x, tid = threadIdx.x;
  hm[tid] = 0.5f * (hn_sa[(size_t)b * kH + tid] +
                    hn_sa[(size_t)(kB + b) * kH + tid]);
  __syncthreads();
  float s = fc1b[tid];
  for (int k = 0; k < kH; ++k) s += fc1w[(size_t)tid * kH + k] * hm[k];
  z1[tid] = tanhf(s);
  __syncthreads();
  if (tid < kJ) {
    float s2 = fc2b[tid];
    for (int k = 0; k < kH; ++k) s2 += fc2w[(size_t)tid * kH + k] * z1[k];
    lg[tid] = s2;
  }
  __syncthreads();
  if (tid == 0) {
    float w[kJ], a[kJ];
    float mx = lg[0];
    for (int j = 1; j < kJ; ++j) mx = fmaxf(mx, lg[j]);
    float sum = 0.f;
    for (int j = 0; j < kJ; ++j) { w[j] = __expf(lg[j] - mx); sum += w[j]; }
    for (int j = 0; j < kJ; ++j) { w[j] /= sum; a[j] = 0.001f; }
    int used[kJ] = {}; float vals[5]; int idx[5];
    for (int k = 0; k < 5; ++k) {         // top-5, lowest index wins ties
      float best = -1.f; int bi = 0;
      for (int j = 0; j < kJ; ++j)
        if (!used[j] && w[j] > best) { best = w[j]; bi = j; }
      used[bi] = 1; vals[k] = best; idx[k] = bi;
    }
    float nrm = 0.f;
    for (int k = 0; k < 5; ++k) if (vals[k] >= 0.1f) nrm += vals[k];
    nrm = fmaxf(nrm, 1e-8f);
    for (int k = 0; k < 5; ++k)
      a[idx[k]] = (vals[k] >= 0.1f) ? vals[k] / nrm : 0.001f;
    for (int j = 0; j < kJ; ++j) alpha[b * kJ + j] = a[j];
  }
}

// ---------------- final FC: out = hn_main @ W^T + b -------------------------

__global__ void k_outfc(const float* __restrict__ hn,     // [6*kB][kH]
                        const float* __restrict__ w,      // [kH][kH]
                        const float* __restrict__ bias,
                        float* __restrict__ out) {
  __shared__ float hrow[kH];
  int row = blockIdx.x;
  hrow[threadIdx.x] = hn[(size_t)row * kH + threadIdx.x];
  __syncthreads();
  float s = bias[threadIdx.x];
  for (int k = 0; k < kH; ++k) s += w[(size_t)threadIdx.x * kH + k] * hrow[k];
  out[(size_t)row * kH + threadIdx.x] = s;
}

// ---------------------------------------------------------------------------

extern "C" void kernel_launch(void* const* d_in, const int* in_sizes, int n_in,
                              void* d_out, int out_size, void* d_ws, size_t ws_size,
                              hipStream_t stream) {
  (void)in_sizes; (void)n_in; (void)out_size; (void)ws_size;
  // d_in order: setup_inputs() dict, params pytree-flattened (sorted keys):
  const float* input   = (const float*)d_in[0];   // [T,B,J]
  const float* h0_sa   = (const float*)d_in[1];   // [2,B,H]
  const float* c0_sa   = (const float*)d_in[2];
  const float* h0_main = (const float*)d_in[3];   // [6,B,H]
  const float* c0_main = (const float*)d_in[4];
  const float* m_bias_b[3] = {(const float*)d_in[5],  (const float*)d_in[11], (const float*)d_in[17]};
  const float* m_bias_f[3] = {(const float*)d_in[6],  (const float*)d_in[12], (const float*)d_in[18]};
  const float* m_whh_b[3]  = {(const float*)d_in[7],  (const float*)d_in[13], (const float*)d_in[19]};
  const float* m_whh_f[3]  = {(const float*)d_in[8],  (const float*)d_in[14], (const float*)d_in[20]};
  const float* m_wih_b[3]  = {(const float*)d_in[9],  (const float*)d_in[15], (const float*)d_in[21]};
  const float* m_wih_f[3]  = {(const float*)d_in[10], (const float*)d_in[16], (const float*)d_in[22]};
  const float* mainfc1_b = (const float*)d_in[23];
  const float* mainfc1_w = (const float*)d_in[24];
  const float* sa_bias_b = (const float*)d_in[25];
  const float* sa_bias_f = (const float*)d_in[26];
  const float* sa_whh_b  = (const float*)d_in[27];
  const float* sa_whh_f  = (const float*)d_in[28];
  const float* sa_wih_b  = (const float*)d_in[29];
  const float* sa_wih_f  = (const float*)d_in[30];
  const float* safc1_b   = (const float*)d_in[31];
  const float* safc1_w   = (const float*)d_in[32];
  const float* safc2_b   = (const float*)d_in[33];
  const float* safc2_w   = (const float*)d_in[34];

  // ---- workspace carve ----
  char* base = (char*)d_ws;
  size_t off = 0;
  auto carve = [&](size_t bytes) -> void* {
    void* p = base + off;
    off = (off + bytes + 255) & ~(size_t)255;
    return p;
  };
  unsigned* bars   = (unsigned*)carve(8 * sizeof(unsigned));
  bf16_t* x25      = (bf16_t*)carve((size_t)kTB * 32 * sizeof(bf16_t));
  bf16_t* yA       = (bf16_t*)carve((size_t)kTB * 2 * kH * sizeof(bf16_t));
  bf16_t* yB       = (bf16_t*)carve((size_t)kTB * 2 * kH * sizeof(bf16_t));
  float*  Gx       = (float*)carve((size_t)2 * kTB * kG * sizeof(float)); // both dirs
  bf16_t* hbuf     = (bf16_t*)carve((size_t)2 * 2 * kB * kH * sizeof(bf16_t));
  float*  hn_sa    = (float*)carve((size_t)2 * kB * kH * sizeof(float));
  float*  hn_main  = (float*)carve((size_t)6 * kB * kH * sizeof(float));
  float*  alpha    = (float*)carve((size_t)kB * kJ * sizeof(float));
  bf16_t* wih_sa[2];  bf16_t* whh_sa[2];
  bf16_t* wih_m[3][2]; bf16_t* whh_m[3][2];   // [layer][dir: 0=f,1=b]
  for (int d = 0; d < 2; ++d) {
    wih_sa[d] = (bf16_t*)carve((size_t)kG * 32 * sizeof(bf16_t));
    whh_sa[d] = (bf16_t*)carve((size_t)kG * kH * sizeof(bf16_t));
  }
  for (int l = 0; l < 3; ++l)
    for (int d = 0; d < 2; ++d) {
      int kpad = (l == 0) ? 32 : 2 * kH;
      wih_m[l][d] = (bf16_t*)carve((size_t)kG * kpad * sizeof(bf16_t));
      whh_m[l][d] = (bf16_t*)carve((size_t)kG * kH * sizeof(bf16_t));
    }

  hipMemsetAsync(bars, 0, 8 * sizeof(unsigned), stream);

  auto cvt = [&](const float* src, bf16_t* dst, int rows, int kin, int kout) {
    int n = rows * kout;
    k_cvt<<<(n + 255) / 256, 256, 0, stream>>>(src, dst, rows, kin, kout);
  };
  // ---- weight conversion ----
  cvt(sa_wih_f, wih_sa[0], kG, kJ, 32);  cvt(sa_wih_b, wih_sa[1], kG, kJ, 32);
  cvt(sa_whh_f, whh_sa[0], kG, kH, kH);  cvt(sa_whh_b, whh_sa[1], kG, kH, kH);
  cvt(m_wih_f[0], wih_m[0][0], kG, kJ, 32);       cvt(m_wih_b[0], wih_m[0][1], kG, kJ, 32);
  cvt(m_wih_f[1], wih_m[1][0], kG, 2 * kH, 2 * kH); cvt(m_wih_b[1], wih_m[1][1], kG, 2 * kH, 2 * kH);
  cvt(m_wih_f[2], wih_m[2][0], kG, 2 * kH, 2 * kH); cvt(m_wih_b[2], wih_m[2][1], kG, 2 * kH, 2 * kH);
  for (int l = 0; l < 3; ++l) {
    cvt(m_whh_f[l], whh_m[l][0], kG, kH, kH);
    cvt(m_whh_b[l], whh_m[l][1], kG, kH, kH);
  }

  const dim3 gemm_grid(kG / 64, kTB / 128, 2);
  const dim3 rec_grid(RNWG, 2);
  const int pack_blocks = (kTB * 32 + 255) / 256;

  // ---- spatial-attention bi-LSTM ----
  k_pack25<<<pack_blocks, 256, 0, stream>>>(input, nullptr, x25);
  k_gates_gemm<<<gemm_grid, 256, 0, stream>>>(x25, wih_sa[0], wih_sa[1],
                                              sa_bias_f, sa_bias_b, Gx, 32);
  k_lstm_recur<<<rec_grid, 256, 0, stream>>>(Gx, whh_sa[0], whh_sa[1],
                                             h0_sa, c0_sa, hbuf, nullptr,
                                             hn_sa, bars + 0);
  k_attn<<<kB, kH, 0, stream>>>(hn_sa, safc1_w, safc1_b, safc2_w, safc2_b, alpha);

  // ---- main layer 0 (input J=25, scaled by alpha) ----
  k_pack25<<<pack_blocks, 256, 0, stream>>>(input, alpha, x25);
  k_gates_gemm<<<gemm_grid, 256, 0, stream>>>(x25, wih_m[0][0], wih_m[0][1],
                                              m_bias_f[0], m_bias_b[0], Gx, 32);
  k_lstm_recur<<<rec_grid, 256, 0, stream>>>(Gx, whh_m[0][0], whh_m[0][1],
                                             h0_main + 0 * kB * kH, c0_main + 0 * kB * kH,
                                             hbuf, yA, hn_main + 0 * kB * kH, bars + 2);
  // ---- main layer 1 (input 2H) ----
  k_gates_gemm<<<gemm_grid, 256, 0, stream>>>(yA, wih_m[1][0], wih_m[1][1],
                                              m_bias_f[1], m_bias_b[1], Gx, 2 * kH);
  k_lstm_recur<<<rec_grid, 256, 0, stream>>>(Gx, whh_m[1][0], whh_m[1][1],
                                             h0_main + 2 * kB * kH, c0_main + 2 * kB * kH,
                                             hbuf, yB, hn_main + 2 * kB * kH, bars + 4);
  // ---- main layer 2 (input 2H, y unused) ----
  k_gates_gemm<<<gemm_grid, 256, 0, stream>>>(yB, wih_m[2][0], wih_m[2][1],
                                              m_bias_f[2], m_bias_b[2], Gx, 2 * kH);
  k_lstm_recur<<<rec_grid, 256, 0, stream>>>(Gx, whh_m[2][0], whh_m[2][1],
                                             h0_main + 4 * kB * kH, c0_main + 4 * kB * kH,
                                             hbuf, nullptr, hn_main + 4 * kB * kH, bars + 6);

  // ---- final FC ----
  k_outfc<<<6 * kB, kH, 0, stream>>>(hn_main, mainfc1_w, mainfc1_b, (float*)d_out);
}
